// SimilarityCalculatorFeatureOnly_49246095016270
// MI455X (gfx1250) — matrix operand order repaired and checked by reading, not compiled, not run on hardware
//
#include <hip/hip_runtime.h>
#include <hip/hip_bf16.h>

// ---------- vector types ----------
typedef float v8f   __attribute__((ext_vector_type(8)));
typedef __bf16 v16bf __attribute__((ext_vector_type(16)));
typedef unsigned int u32x4 __attribute__((ext_vector_type(4)));
typedef unsigned int u32x8 __attribute__((ext_vector_type(8)));
typedef unsigned short u16x4 __attribute__((ext_vector_type(4)));

#define TILE      128
#define TILE_K    32
#define LDS_STR   40                   // ushorts per row: 32 data + 8 pad (80B)
#define ARR_USH   (TILE * LDS_STR)     // 5120 ushorts per tile plane
#define ARR_BYTES (ARR_USH * 2)        // 10240 bytes per tile plane
#define D_DIM     512

// ---------- fp32 -> bf16 (round to nearest even) + hi/lo split ----------
static __device__ __forceinline__ unsigned short bf16_rne(float f) {
    unsigned int u = __float_as_uint(f);
    u += 0x7FFFu + ((u >> 16) & 1u);
    return (unsigned short)(u >> 16);
}
static __device__ __forceinline__ float bf16_to_f32(unsigned short h) {
    return __uint_as_float(((unsigned int)h) << 16);
}
struct HL { unsigned short h, l; };
static __device__ __forceinline__ HL cvt_split(float f) {
    HL r;
    r.h = bf16_rne(f);
    r.l = bf16_rne(f - bf16_to_f32(r.h));
    return r;
}

// ---------- gfx1250 async global->LDS copy (ASYNCcnt-tracked) ----------
static __device__ __forceinline__ void async_copy16(unsigned lds_addr,
                                                    const unsigned short* gbase,
                                                    unsigned byte_off) {
#if defined(__HIP_DEVICE_COMPILE__)
    asm volatile("global_load_async_to_lds_b128 %0, %1, %2 offset:0"
                 :: "v"(lds_addr), "v"(byte_off), "s"(gbase) : "memory");
#endif
}
static __device__ __forceinline__ void wait_async0() {
#if defined(__HIP_DEVICE_COMPILE__)
    asm volatile("s_wait_asynccnt 0x0" ::: "memory");
#endif
}

// ---------- fragment assembly from LDS (per ISA 16-bit layouts) ----------
static __device__ __forceinline__ v16bf load_a_frag(const unsigned short* rowp, int hh) {
    u32x4 c0 = *(const u32x4*)(rowp + 8 * hh);
    u32x4 c1 = *(const u32x4*)(rowp + 16 + 8 * hh);
    u32x8 u;
    u[0]=c0[0]; u[1]=c0[1]; u[2]=c0[2]; u[3]=c0[3];
    u[4]=c1[0]; u[5]=c1[1]; u[6]=c1[2]; u[7]=c1[3];
    return __builtin_bit_cast(v16bf, u);
}
static __device__ __forceinline__ v16bf load_b_frag(const unsigned short* rowp, int hh) {
    u32x4 c0 = *(const u32x4*)(rowp + 8 * hh);
    u32x4 c1 = *(const u32x4*)(rowp + 16 + 8 * hh);
    u32x8 u;
    u[0]=c0[0]; u[1]=c0[1]; u[2]=c0[2]; u[3]=c0[3];
    u[4]=c1[0]; u[5]=c1[1]; u[6]=c1[2]; u[7]=c1[3];
    return __builtin_bit_cast(v16bf, u);
}

static __device__ __forceinline__ v8f wmma_bf16(v16bf a, v16bf b, v8f c) {
    return __builtin_amdgcn_wmma_f32_16x16x32_bf16(
        false, a, false, b, (short)0, c, false, false);
}

// =====================================================================
// Pre-pass: fused bf16 hi/lo split + row squared-norm (+ optional copy)
// one wave32 per row
// =====================================================================
__global__ __launch_bounds__(256) void split_norm_kernel(
    const float* __restrict__ in, unsigned short* __restrict__ hi,
    unsigned short* __restrict__ lo, float* __restrict__ norm,
    float* __restrict__ passthru, int rows)
{
    int row  = blockIdx.x * 8 + (threadIdx.x >> 5);
    int lane = threadIdx.x & 31;
    if (row >= rows) return;
    const float4* p = (const float4*)(in + (size_t)row * D_DIM);
    float s = 0.0f;
    #pragma unroll
    for (int i = 0; i < 4; ++i) {
        int idx = lane + 32 * i;
        float4 v = p[idx];
        if (passthru) ((float4*)(passthru + (size_t)row * D_DIM))[idx] = v;
        u16x4 h4, l4; HL t;
        t = cvt_split(v.x); h4[0]=t.h; l4[0]=t.l;
        t = cvt_split(v.y); h4[1]=t.h; l4[1]=t.l;
        t = cvt_split(v.z); h4[2]=t.h; l4[2]=t.l;
        t = cvt_split(v.w); h4[3]=t.h; l4[3]=t.l;
        *(u16x4*)(hi + (size_t)row * D_DIM + idx * 4) = h4;
        *(u16x4*)(lo + (size_t)row * D_DIM + idx * 4) = l4;
        s += v.x*v.x + v.y*v.y + v.z*v.z + v.w*v.w;
    }
    #pragma unroll
    for (int off = 16; off > 0; off >>= 1) s += __shfl_xor(s, off, 32);
    if (lane == 0) norm[row] = s;
}

// ---------- row squared-norms (fallback path) ----------
__global__ __launch_bounds__(256) void row_norms_kernel(const float* __restrict__ x,
                                                        float* __restrict__ out, int rows) {
    int row  = blockIdx.x * 8 + (threadIdx.x >> 5);
    int lane = threadIdx.x & 31;
    if (row >= rows) return;
    const float4* p = (const float4*)(x + (size_t)row * D_DIM);
    float s = 0.0f;
    #pragma unroll
    for (int i = 0; i < 4; ++i) {
        float4 v = p[lane + 32 * i];
        s += v.x*v.x + v.y*v.y + v.z*v.z + v.w*v.w;
    }
    #pragma unroll
    for (int off = 16; off > 0; off >>= 1) s += __shfl_xor(s, off, 32);
    if (lane == 0) out[row] = s;
}

__global__ __launch_bounds__(256) void copy4_kernel(const float* __restrict__ in,
                                                    float* __restrict__ out, int n4) {
    int i = blockIdx.x * blockDim.x + threadIdx.x;
    if (i < n4) ((float4*)out)[i] = ((const float4*)in)[i];
}

// =====================================================================
// Fast GEMM: async-to-LDS double-buffered staging of pre-split bf16
// planes + 3-pass bf16 WMMA + fused similarity epilogue
// =====================================================================
__global__ __launch_bounds__(256) void gemm_sim_async_kernel(
    const unsigned short* __restrict__ bHi, const unsigned short* __restrict__ bLo,
    const unsigned short* __restrict__ tHi, const unsigned short* __restrict__ tLo,
    const float* __restrict__ x2, const float* __restrict__ y2,
    float* __restrict__ sim, int N, int M)
{
    // lds[buf][plane][row*LDS_STR + k]; planes: 0=Ahi 1=Alo 2=Bhi 3=Blo
    __shared__ unsigned short lds[2][4][ARR_USH];

    const int tid   = threadIdx.x;
    const int lane  = tid & 31;
    const int wave  = tid >> 5;
    const int waveM = wave & 3;
    const int waveN = wave >> 2;
    const int blockRow = blockIdx.y * TILE;
    const int blockCol = blockIdx.x * TILE;
    const int m  = lane & 15;
    const int hh = lane >> 4;

    v8f acc[2][4];
    #pragma unroll
    for (int rt = 0; rt < 2; ++rt)
        #pragma unroll
        for (int ct = 0; ct < 4; ++ct)
            acc[rt][ct] = (v8f){0.f,0.f,0.f,0.f,0.f,0.f,0.f,0.f};

    // issue 8 async b128 copies (2 chunks x 4 planes) for K-step k0 into buf
    auto prefetch = [&](int buf, int k0) {
        #pragma unroll
        for (int j = 0; j < 2; ++j) {
            int chunk = tid + 256 * j;         // 0..511 : row = chunk>>2, c = chunk&3
            int row = chunk >> 2;
            int c   = chunk & 3;
            unsigned aoff = (unsigned)(((blockRow + row) * D_DIM + k0 + c * 8) * 2);
            unsigned boff = (unsigned)(((blockCol + row) * D_DIM + k0 + c * 8) * 2);
            unsigned la = (unsigned)(uintptr_t)&lds[buf][0][row * LDS_STR + c * 8];
            async_copy16(la,                 bHi, aoff);
            async_copy16(la + ARR_BYTES,     bLo, aoff);
            async_copy16(la + 2 * ARR_BYTES, tHi, boff);
            async_copy16(la + 3 * ARR_BYTES, tLo, boff);
        }
    };

    int cur = 0;
    prefetch(0, 0);

    for (int ks = 0; ks < D_DIM / TILE_K; ++ks) {
        wait_async0();
        __syncthreads();
        if (ks + 1 < D_DIM / TILE_K) prefetch(cur ^ 1, (ks + 1) * TILE_K);

        const unsigned short* Ahi = &lds[cur][0][0];
        const unsigned short* Alo = &lds[cur][1][0];
        const unsigned short* Bhi = &lds[cur][2][0];
        const unsigned short* Blo = &lds[cur][3][0];

        const int arow0 = (waveM * 32 + m) * LDS_STR;
        const int arow1 = (waveM * 32 + 16 + m) * LDS_STR;
        v16bf ah0 = load_a_frag(Ahi + arow0, hh);
        v16bf al0 = load_a_frag(Alo + arow0, hh);
        v16bf ah1 = load_a_frag(Ahi + arow1, hh);
        v16bf al1 = load_a_frag(Alo + arow1, hh);

        #pragma unroll
        for (int ct = 0; ct < 4; ++ct) {
            const int brow = (waveN * 64 + ct * 16 + m) * LDS_STR;
            v16bf bh = load_b_frag(Bhi + brow, hh);
            v16bf bl = load_b_frag(Blo + brow, hh);
            acc[0][ct] = wmma_bf16(ah0, bh, acc[0][ct]);
            acc[0][ct] = wmma_bf16(ah0, bl, acc[0][ct]);
            acc[0][ct] = wmma_bf16(al0, bh, acc[0][ct]);
            acc[1][ct] = wmma_bf16(ah1, bh, acc[1][ct]);
            acc[1][ct] = wmma_bf16(ah1, bl, acc[1][ct]);
            acc[1][ct] = wmma_bf16(al1, bh, acc[1][ct]);
        }
        cur ^= 1;
    }

    // epilogue: sq = max(x2+y2-2*xy, 0); sim = min(1/sqrt(sq), 1e6)
    #pragma unroll
    for (int rt = 0; rt < 2; ++rt) {
        const int row0 = blockRow + waveM * 32 + rt * 16 + hh * 8;
        float xn[8];
        #pragma unroll
        for (int r = 0; r < 8; ++r) xn[r] = x2[row0 + r];
        #pragma unroll
        for (int ct = 0; ct < 4; ++ct) {
            const int col = blockCol + waveN * 64 + ct * 16 + m;
            const float yn = y2[col];
            #pragma unroll
            for (int r = 0; r < 8; ++r) {
                float sq = fmaxf(xn[r] + yn - 2.0f * acc[rt][ct][r], 0.0f);
                float s  = fminf(1.0f / sqrtf(sq), 1.0e6f);
                sim[(size_t)(row0 + r) * (size_t)M + (size_t)col] = s;
            }
        }
    }
}

// =====================================================================
// Fallback GEMM (convert-in-kernel, no workspace planes needed)
// =====================================================================
__global__ __launch_bounds__(256) void gemm_sim_kernel(
    const float* __restrict__ base, const float* __restrict__ tgt,
    const float* __restrict__ x2, const float* __restrict__ y2,
    float* __restrict__ sim, int N, int M)
{
    __shared__ unsigned short ldsAhi[ARR_USH];
    __shared__ unsigned short ldsAlo[ARR_USH];
    __shared__ unsigned short ldsBhi[ARR_USH];
    __shared__ unsigned short ldsBlo[ARR_USH];

    const int tid   = threadIdx.x;
    const int lane  = tid & 31;
    const int wave  = tid >> 5;
    const int waveM = wave & 3;
    const int waveN = wave >> 2;
    const int blockRow = blockIdx.y * TILE;
    const int blockCol = blockIdx.x * TILE;
    const int m  = lane & 15;
    const int hh = lane >> 4;

    v8f acc[2][4];
    #pragma unroll
    for (int rt = 0; rt < 2; ++rt)
        #pragma unroll
        for (int ct = 0; ct < 4; ++ct)
            acc[rt][ct] = (v8f){0.f,0.f,0.f,0.f,0.f,0.f,0.f,0.f};

    const int srow = tid >> 3;
    const int skq  = tid & 7;

    for (int k0 = 0; k0 < D_DIM; k0 += TILE_K) {
        __syncthreads();
        #pragma unroll
        for (int i = 0; i < 4; ++i) {
            int row = i * 32 + srow;
            float4 va = *(const float4*)(base + (size_t)(blockRow + row) * D_DIM + k0 + skq * 4);
            float4 vb = *(const float4*)(tgt  + (size_t)(blockCol + row) * D_DIM + k0 + skq * 4);
            u16x4 ah, al, bh, bl; HL t;
            t = cvt_split(va.x); ah[0]=t.h; al[0]=t.l;
            t = cvt_split(va.y); ah[1]=t.h; al[1]=t.l;
            t = cvt_split(va.z); ah[2]=t.h; al[2]=t.l;
            t = cvt_split(va.w); ah[3]=t.h; al[3]=t.l;
            t = cvt_split(vb.x); bh[0]=t.h; bl[0]=t.l;
            t = cvt_split(vb.y); bh[1]=t.h; bl[1]=t.l;
            t = cvt_split(vb.z); bh[2]=t.h; bl[2]=t.l;
            t = cvt_split(vb.w); bh[3]=t.h; bl[3]=t.l;
            int o = row * LDS_STR + skq * 4;
            *(u16x4*)(ldsAhi + o) = ah;
            *(u16x4*)(ldsAlo + o) = al;
            *(u16x4*)(ldsBhi + o) = bh;
            *(u16x4*)(ldsBlo + o) = bl;
        }
        __syncthreads();

        const int arow0 = (waveM * 32 + m) * LDS_STR;
        const int arow1 = (waveM * 32 + 16 + m) * LDS_STR;
        v16bf ah0 = load_a_frag(ldsAhi + arow0, hh);
        v16bf al0 = load_a_frag(ldsAlo + arow0, hh);
        v16bf ah1 = load_a_frag(ldsAhi + arow1, hh);
        v16bf al1 = load_a_frag(ldsAlo + arow1, hh);

        #pragma unroll
        for (int ct = 0; ct < 4; ++ct) {
            const int brow = (waveN * 64 + ct * 16 + m) * LDS_STR;
            v16bf bh = load_b_frag(ldsBhi + brow, hh);
            v16bf bl = load_b_frag(ldsBlo + brow, hh);
            acc[0][ct] = wmma_bf16(ah0, bh, acc[0][ct]);
            acc[0][ct] = wmma_bf16(ah0, bl, acc[0][ct]);
            acc[0][ct] = wmma_bf16(al0, bh, acc[0][ct]);
            acc[1][ct] = wmma_bf16(ah1, bh, acc[1][ct]);
            acc[1][ct] = wmma_bf16(ah1, bl, acc[1][ct]);
            acc[1][ct] = wmma_bf16(al1, bh, acc[1][ct]);
        }
    }

    #pragma unroll
    for (int rt = 0; rt < 2; ++rt) {
        const int row0 = blockRow + waveM * 32 + rt * 16 + hh * 8;
        float xn[8];
        #pragma unroll
        for (int r = 0; r < 8; ++r) xn[r] = x2[row0 + r];
        #pragma unroll
        for (int ct = 0; ct < 4; ++ct) {
            const int col = blockCol + waveN * 64 + ct * 16 + m;
            const float yn = y2[col];
            #pragma unroll
            for (int r = 0; r < 8; ++r) {
                float sq = fmaxf(xn[r] + yn - 2.0f * acc[rt][ct][r], 0.0f);
                float s  = fminf(1.0f / sqrtf(sq), 1.0e6f);
                sim[(size_t)(row0 + r) * (size_t)M + (size_t)col] = s;
            }
        }
    }
}

extern "C" void kernel_launch(void* const* d_in, const int* in_sizes, int n_in,
                              void* d_out, int out_size, void* d_ws, size_t ws_size,
                              hipStream_t stream) {
    const float* base = (const float*)d_in[0];
    const float* tgt  = (const float*)d_in[1];
    const int N = in_sizes[0] / D_DIM;   // 4096
    const int M = in_sizes[1] / D_DIM;   // 8192

    float* out_base = (float*)d_out;
    float* out_sim  = out_base + (size_t)N * D_DIM;
    dim3 grid(M / TILE, N / TILE);

    // fast path needs hi/lo bf16 planes for both matrices + norms
    const size_t planeBytes = (size_t)(N + M) * D_DIM * 2 /*hi+lo*/ * sizeof(unsigned short);
    const size_t needFast   = planeBytes + (size_t)(N + M) * sizeof(float);

    if (ws_size >= needFast) {
        unsigned short* bHi = (unsigned short*)d_ws;
        unsigned short* bLo = bHi + (size_t)N * D_DIM;
        unsigned short* tHi = bLo + (size_t)N * D_DIM;
        unsigned short* tLo = tHi + (size_t)M * D_DIM;
        float* x2 = (float*)(tLo + (size_t)M * D_DIM);
        float* y2 = x2 + N;

        // fused: split to bf16 hi/lo + norms (+ passthrough copy for base)
        split_norm_kernel<<<(N + 7) / 8, 256, 0, stream>>>(base, bHi, bLo, x2, out_base, N);
        split_norm_kernel<<<(M + 7) / 8, 256, 0, stream>>>(tgt, tHi, tLo, y2, nullptr, M);

        gemm_sim_async_kernel<<<grid, 256, 0, stream>>>(bHi, bLo, tHi, tLo,
                                                        x2, y2, out_sim, N, M);
    } else {
        float* x2 = (float*)d_ws;
        float* y2 = x2 + N;
        int n4 = (N * D_DIM) / 4;
        copy4_kernel<<<(n4 + 255) / 256, 256, 0, stream>>>(base, out_base, n4);
        row_norms_kernel<<<(N + 7) / 8, 256, 0, stream>>>(base, x2, N);
        row_norms_kernel<<<(M + 7) / 8, 256, 0, stream>>>(tgt, y2, M);
        gemm_sim_kernel<<<grid, 256, 0, stream>>>(base, tgt, x2, y2, out_sim, N, M);
    }
}
